// VGAE_ENCODER_89567247991228
// MI455X (gfx1250) — matrix operand order repaired and checked by reading, not compiled, not run on hardware
//
#include <hip/hip_runtime.h>
#include <stdint.h>

typedef __attribute__((ext_vector_type(16))) __bf16 v16bf;
typedef __attribute__((ext_vector_type(8)))  __bf16 v8bf;
typedef __attribute__((ext_vector_type(8)))  float  v8f;

#define N_NODES 16384
#define IN_DIM  256
#define H1      128
#define H2      64

// f32 -> bf16 round-to-nearest (half-up): scalar helper for the small kernels.
__device__ __forceinline__ unsigned short f2bf(float f) {
  unsigned u = __builtin_bit_cast(unsigned, f) + 0x8000u;
  return (unsigned short)(u >> 16);
}

// Pack two f32 -> two bf16 (round-half-up) in 3 full-rate VALU ops:
// 2x v_add_nc_u32 + 1x v_perm_b32 (bytes {y3,y2,x3,x2}).
__device__ __forceinline__ uint32_t pack2_bf16(float x, float y) {
  uint32_t ux = __builtin_bit_cast(uint32_t, x) + 0x8000u;
  uint32_t uy = __builtin_bit_cast(uint32_t, y) + 0x8000u;
  return __builtin_amdgcn_perm(uy, ux, 0x07060302u);
}

// ---------------------------------------------------------------------------
// C[M,128] = act( adj[M,16384] (f32) @ B[16384,128] (bf16 raw) ), C in f32.
// WG = 256 threads (8 waves), 64-row x 128-col output tile, KT=64 stages.
// Double-buffered pipeline: B tile prefetched async global->LDS (ASYNCcnt),
// A tile (f32) prefetched into registers, converted to bf16 on the LDS-store
// path. adj traffic is the roofline term: 1 GiB streamed once per GEMM.
// ---------------------------------------------------------------------------
template <bool RELU>
__global__ __launch_bounds__(256) void k_gemm_adj(const float* __restrict__ adj,
                                                  const unsigned short* __restrict__ Bg,
                                                  float* __restrict__ C) {
  constexpr int MT = 64;   // rows per WG
  constexpr int KT = 64;   // K per LDS stage
  constexpr int AS = 72;   // A LDS row stride in bf16 elems (pad: bank spread)
  constexpr int BS = 136;  // B LDS row stride in bf16 elems (pad: bank spread)
  constexpr uint32_t A_BYTES = MT * AS * 2;               // 9216
  constexpr uint32_t B_BYTES = KT * BS * 2;               // 17408
  constexpr uint32_t AOFF[2] = {0u, A_BYTES};             // A double buffer
  constexpr uint32_t BOFF[2] = {2 * A_BYTES, 2 * A_BYTES + B_BYTES};
  __shared__ __align__(16) unsigned char smem[2 * A_BYTES + 2 * B_BYTES];

  const int tid = threadIdx.x;
  const int lane = tid & 31;
  const int wave = tid >> 5;
  const int lane_lo = lane & 15;
  const int hi = (lane >> 4) & 1;
  const int mslab = wave & 3;   // which 16-row slab
  const int npair = wave >> 2;  // which 64-col half
  const int m0 = blockIdx.x * MT;

  const uint32_t smem_lds = (uint32_t)(uintptr_t)smem;  // wave-relative LDS addr

  // Per-thread tile coordinates for the cooperative loads (4 chunks each).
  int a_r[4], a_cv[4];
#pragma unroll
  for (int c = 0; c < 4; ++c) {
    int idx = c * 256 + tid;
    a_r[c] = idx >> 4;   // 0..63
    a_cv[c] = idx & 15;  // 0..15
  }

  // ---- prologue: stage 0 in flight -------------------------------------
  // B stage 0: async global->LDS, 16B per lane per op, 4 ops/thread.
#pragma unroll
  for (int c = 0; c < 4; ++c) {
    uint32_t voff = (uint32_t)(a_r[c] * 256 + a_cv[c] * 16);
    uint32_t laddr = smem_lds + BOFF[0] + (uint32_t)(a_r[c] * (BS * 2) + a_cv[c] * 16);
    asm volatile("global_load_async_to_lds_b128 %0, %1, %2"
                 :: "v"(laddr), "v"(voff), "s"(Bg) : "memory");
  }
  // A stage 0: f32 into registers.
  float4 areg[4];
#pragma unroll
  for (int c = 0; c < 4; ++c)
    areg[c] = *(const float4*)(adj + (size_t)(m0 + a_r[c]) * N_NODES + a_cv[c] * 4);

  v8f acc[4] = {v8f{}, v8f{}, v8f{}, v8f{}};
  int p = 0;

  for (int kt = 0; kt < N_NODES; kt += KT) {
    // ---- convert + store A(stage) regs -> LDS buf p
    const uint32_t apo = p ? AOFF[1] : AOFF[0];
    const uint32_t bpo = p ? BOFF[1] : BOFF[0];
#pragma unroll
    for (int c = 0; c < 4; ++c) {
      uint2 pk;
      pk.x = pack2_bf16(areg[c].x, areg[c].y);
      pk.y = pack2_bf16(areg[c].z, areg[c].w);
      *(uint2*)(smem + apo + a_r[c] * (AS * 2) + a_cv[c] * 8) = pk;
    }

    // ---- prefetch next stage while this one computes
    const int ktn = kt + KT;
    if (ktn < N_NODES) {
      const unsigned short* bsrc = Bg + (size_t)ktn * H1;
      const uint32_t bno = p ? BOFF[0] : BOFF[1];
#pragma unroll
      for (int c = 0; c < 4; ++c) {
        uint32_t voff = (uint32_t)(a_r[c] * 256 + a_cv[c] * 16);
        uint32_t laddr = smem_lds + bno + (uint32_t)(a_r[c] * (BS * 2) + a_cv[c] * 16);
        asm volatile("global_load_async_to_lds_b128 %0, %1, %2"
                     :: "v"(laddr), "v"(voff), "s"(bsrc) : "memory");
      }
#pragma unroll
      for (int c = 0; c < 4; ++c)
        areg[c] = *(const float4*)(adj + (size_t)(m0 + a_r[c]) * N_NODES + ktn + a_cv[c] * 4);
      // Async ops complete in order: <=4 outstanding => current stage landed.
      asm volatile("s_wait_asynccnt 0x4" ::: "memory");
    } else {
      asm volatile("s_wait_asynccnt 0x0" ::: "memory");
    }
    __syncthreads();  // stage data visible to all waves

    // ---- compute: 2 K-subtiles of 32; A reused across the wave's 4 N-tiles
#pragma unroll
    for (int kk = 0; kk < KT; kk += 32) {
      // A 16x32 frag: lane<16 holds M=lane_lo, K={0..7,16..23}; hi half +8/+24
      const unsigned char* ap =
          smem + apo + (mslab * 16 + lane_lo) * (AS * 2) + kk * 2 + hi * 16;
      v8bf alo = *(const v8bf*)(ap);
      v8bf ahi = *(const v8bf*)(ap + 32);
      v16bf A;
#pragma unroll
      for (int i = 0; i < 8; ++i) { A[i] = alo[i]; A[i + 8] = ahi[i]; }

      // Preload all 4 B frags (lane holds col N=lane_lo, K={0..7,16..23}; hi half +8/+24)
      const unsigned char* bp0 =
          smem + bpo + (kk + hi * 8) * (BS * 2) + npair * 128 + lane_lo * 2;
      v8bf bl[4], bh[4];
#pragma unroll
      for (int t = 0; t < 4; ++t) {
#pragma unroll
        for (int i = 0; i < 8; ++i) {
          bl[t][i] = *(const __bf16*)(bp0 + i * (BS * 2) + t * 32);
          bh[t][i] = *(const __bf16*)(bp0 + (i + 16) * (BS * 2) + t * 32);
        }
      }
#pragma unroll
      for (int t = 0; t < 4; ++t) {
        v16bf B;
#pragma unroll
        for (int i = 0; i < 8; ++i) { B[i] = bl[t][i]; B[i + 8] = bh[t][i]; }
        acc[t] = __builtin_amdgcn_wmma_f32_16x16x32_bf16(
            false, A, false, B, (short)0, acc[t], false, false);
      }
    }
    __syncthreads();  // all waves done reading buf p before it is refilled
    p ^= 1;
  }

  // Epilogue: C layout lane<16 -> N=lane, M=vgpr; lane>=16 -> M=8+vgpr
  const int mrow0 = m0 + mslab * 16 + hi * 8;
#pragma unroll
  for (int t = 0; t < 4; ++t) {
    const int n = (npair * 4 + t) * 16 + lane_lo;
#pragma unroll
    for (int i = 0; i < 8; ++i) {
      float v = acc[t][i];
      if (RELU) v = fmaxf(v, 0.f);
      C[(size_t)(mrow0 + i) * H1 + n] = v;
    }
  }
}

// ---------------------------------------------------------------------------
// B1 = bf16(X @ W1): one row per block, 128 outputs. ~1 GFLOP, VALU is fine.
// ---------------------------------------------------------------------------
__global__ __launch_bounds__(128) void k_xw1(const float* __restrict__ X,
                                             const float* __restrict__ W1,
                                             unsigned short* __restrict__ B1) {
  __shared__ float xrow[IN_DIM];
  const int r = blockIdx.x, c = threadIdx.x;
  xrow[c] = X[(size_t)r * IN_DIM + c];
  xrow[c + 128] = X[(size_t)r * IN_DIM + 128 + c];
  __syncthreads();
  float s = 0.f;
#pragma unroll 8
  for (int k = 0; k < IN_DIM; ++k) s = fmaf(xrow[k], W1[k * H1 + c], s);
  B1[(size_t)r * H1 + c] = f2bf(s);
}

// ---------------------------------------------------------------------------
// T = bf16(hidden @ [Wm | Ws])  — fuses the mean/logstd branches' 2nd GCN
// into one 128-col B operand for a single adj-GEMM pass.
// ---------------------------------------------------------------------------
__global__ __launch_bounds__(128) void k_t(const float* __restrict__ Hd,
                                           const float* __restrict__ Wm,
                                           const float* __restrict__ Ws,
                                           unsigned short* __restrict__ T) {
  __shared__ float hrow[H1];
  const int r = blockIdx.x, c = threadIdx.x;
  hrow[c] = Hd[(size_t)r * H1 + c];
  __syncthreads();
  const float* W = (c < H2) ? Wm : Ws;
  const int cc = (c < H2) ? c : c - H2;
  float s = 0.f;
#pragma unroll 8
  for (int k = 0; k < H1; ++k) s = fmaf(hrow[k], W[k * H2 + cc], s);
  T[(size_t)r * H1 + c] = f2bf(s);
}

// ---------------------------------------------------------------------------
// out = noise * exp(min(logstd,10)) + Z@Wp + bp
// ZS[:, :64] = Z, ZS[:, 64:] = logstd.
// ---------------------------------------------------------------------------
__global__ __launch_bounds__(64) void k_final(const float* __restrict__ ZS,
                                              const float* __restrict__ Wp,
                                              const float* __restrict__ bp,
                                              const float* __restrict__ noise,
                                              float* __restrict__ out) {
  __shared__ float zrow[H2];
  const int r = blockIdx.x, j = threadIdx.x;
  zrow[j] = ZS[(size_t)r * H1 + j];
  __syncthreads();
  float mean = bp[j];
#pragma unroll 8
  for (int h = 0; h < H2; ++h) mean = fmaf(zrow[h], Wp[h * H2 + j], mean);
  float ls = fminf(ZS[(size_t)r * H1 + H2 + j], 10.f);
  out[(size_t)r * H2 + j] = noise[(size_t)r * H2 + j] * expf(ls) + mean;
}

extern "C" void kernel_launch(void* const* d_in, const int* in_sizes, int n_in,
                              void* d_out, int out_size, void* d_ws, size_t ws_size,
                              hipStream_t stream) {
  (void)in_sizes; (void)n_in; (void)out_size; (void)ws_size;
  const float* X     = (const float*)d_in[0];
  const float* adj   = (const float*)d_in[1];
  const float* W1    = (const float*)d_in[2];
  const float* Wm    = (const float*)d_in[3];
  const float* Ws    = (const float*)d_in[4];
  const float* Wp    = (const float*)d_in[5];
  const float* bp    = (const float*)d_in[6];
  const float* noise = (const float*)d_in[7];

  char* ws = (char*)d_ws;                                   // 24 MiB used
  unsigned short* B1 = (unsigned short*)(ws);               //  4 MiB bf16
  float* hidden      = (float*)(ws + (4u << 20));           //  8 MiB f32
  unsigned short* T  = (unsigned short*)(ws + (12u << 20)); //  4 MiB bf16
  float* ZS          = (float*)(ws + (16u << 20));          //  8 MiB f32

  k_xw1<<<N_NODES, 128, 0, stream>>>(X, W1, B1);
  k_gemm_adj<true><<<N_NODES / 64, 256, 0, stream>>>(adj, B1, hidden);
  k_t<<<N_NODES, 128, 0, stream>>>(hidden, Wm, Ws, T);
  k_gemm_adj<false><<<N_NODES / 64, 256, 0, stream>>>(adj, T, ZS);
  k_final<<<N_NODES, 64, 0, stream>>>(ZS, Wp, bp, noise, (float*)d_out);
}
